// ODE_RNN_22909355557489
// MI455X (gfx1250) — matrix operand order repaired and checked by reading, not compiled, not run on hardware
//
#include <hip/hip_runtime.h>
#include <hip/hip_bf16.h>
#include <math.h>

typedef __attribute__((ext_vector_type(16))) _Float16 v16h;
typedef __attribute__((ext_vector_type(8)))  _Float16 v8h;
typedef __attribute__((ext_vector_type(8)))  float    v8f;

namespace {
constexpr int LO_    = 64;
constexpr int B_     = 8192;
constexpr int T_     = 256;
constexpr int NSTEP  = 8;
constexpr float THOR = 5.0f;

constexpr int BTILE    = 128;          // batch rows per workgroup
constexpr int NWAVE    = BTILE / 16;   // 8 waves
constexpr int NTHREADS = NWAVE * 32;   // 256 threads
constexpr int STR      = 136;          // staging row stride in halfs (bank spread)

// ---- LDS layout: f16 region (offsets in halfs) ----
constexpr int H_W1t   = 0;                     // 128 x 64   (N x K)
constexpr int H_W2t   = H_W1t  + 128 * 64;     // 128 x 128
constexpr int H_W3t   = H_W2t  + 128 * 128;    // 64  x 128
constexpr int H_Wu1t  = H_W3t  + 64  * 128;    // 128 x 128
constexpr int H_Wu2t  = H_Wu1t + 128 * 128;    // 64  x 128
constexpr int H_Wr1t  = H_Wu2t + 64  * 128;    // 128 x 128
constexpr int H_Wr2t  = H_Wr1t + 128 * 128;    // 64  x 128
constexpr int H_Wn1t  = H_Wr2t + 64  * 128;    // 128 x 128
constexpr int H_Wn2t  = H_Wn1t + 128 * 128;    // 128 x 128
constexpr int H_STAGE = H_Wn2t + 128 * 128;    // per-wave 2 x (16 x STR)
constexpr int H_TOTAL = H_STAGE + NWAVE * 2 * 16 * STR;

// ---- LDS layout: f32 region (offsets in floats, after f16 region) ----
constexpr int F_b1 = 0,  F_b2 = 128, F_b3 = 256, F_bu1 = 320, F_bu2 = 448,
              F_br1 = 512, F_br2 = 640, F_bn1 = 704, F_bn2 = 832,
              F_wu1l = 960, F_wr1l = 1088, F_wn1l = 1216,
              F_xls = 1344, F_mls = F_xls + NWAVE * 16,
              F_TOTAL = F_mls + NWAVE * 16;

constexpr size_t SMEM_BYTES = (size_t)H_TOTAL * 2 + (size_t)F_TOTAL * 4;
} // namespace

// ---- fast transcendentals: use gfx1250 v_tanh_f32 when available ----
__device__ __forceinline__ float fast_rcp(float x) {
#if __has_builtin(__builtin_amdgcn_rcpf)
  return __builtin_amdgcn_rcpf(x);
#else
  return 1.0f / x;
#endif
}
__device__ __forceinline__ float fast_tanh(float x) {
#if __has_builtin(__builtin_amdgcn_tanhf)
  return __builtin_amdgcn_tanhf(x);
#else
  // tanh(x) = 1 - 2 / (exp(2x) + 1)
  float e = __expf(2.0f * x);
  return 1.0f - 2.0f * fast_rcp(e + 1.0f);
#endif
}
__device__ __forceinline__ float fast_sigmoid(float x) {
  return 0.5f * fast_tanh(0.5f * x) + 0.5f;
}

__device__ __forceinline__ v8f wmma_16x16x32(v16h a, v16h b, v8f c) {
  return __builtin_amdgcn_wmma_f32_16x16x32_f16(false, a, false, b,
                                                (short)0, c, false, false);
}

__device__ __forceinline__ v8f zero8() {
  v8f z;
#pragma unroll
  for (int r = 0; r < 8; ++r) z[r] = 0.0f;
  return z;
}

// A-fragment (16x32 f16, M across lanes, K packed in VGPRs) from a row-major
// 16 x STR staging buffer. ISA layout: lanes 0-15 row=l, K k0+0..7 / k0+16..23;
// lanes 16-31 row=l-16, K k0+8..15 / k0+24..31.
__device__ __forceinline__ v16h load_A_frag(const _Float16* act, int k0) {
  const int l   = threadIdx.x & 31;
  const int row = l & 15;
  const int kb  = k0 + ((l & 16) ? 8 : 0);
  const _Float16* p = act + row * STR + kb;
  v8h lo = *(const v8h*)(p);
  v8h hi = *(const v8h*)(p + 16);
  v16h a;
#pragma unroll
  for (int i = 0; i < 8; ++i) { a[i] = lo[i]; a[i + 8] = hi[i]; }
  return a;
}

// Hoist the whole A-operand (K up to 128) once per layer; it is invariant
// across all N-tiles, halving LDS reads per WMMA.
template <int KC>
__device__ __forceinline__ void load_A_set(const _Float16* act, v16h af[KC]) {
#pragma unroll
  for (int kc = 0; kc < KC; ++kc) af[kc] = load_A_frag(act, kc * 32);
}

// B-fragment (32x16 f16) from transposed weight stored N rows x K cols.
// Lanes 0-15: col=l, K k0+0..7 / k0+16..23 ; lanes 16-31: col=l-16, K k0+8..15 / k0+24..31.
__device__ __forceinline__ v16h load_B_frag(const _Float16* wt, int stride,
                                            int k0, int ntile) {
  const int l  = threadIdx.x & 31;
  const int n  = ntile * 16 + (l & 15);
  const int kb = k0 + ((l & 16) ? 8 : 0);
  const _Float16* p = wt + n * stride + kb;
  v8h lo = *(const v8h*)(p);
  v8h hi = *(const v8h*)(p + 16);
  v16h b;
#pragma unroll
  for (int i = 0; i < 8; ++i) { b[i] = lo[i]; b[i + 8] = hi[i]; }
  return b;
}

// One 16x16 output tile with preloaded A-set.
template <int KC>
__device__ __forceinline__ v8f gemm_preA(const v16h af[KC], const _Float16* wt,
                                         int wstride, int ntile) {
  v8f acc = zero8();
#pragma unroll
  for (int kc = 0; kc < KC; ++kc)
    acc = wmma_16x16x32(af[kc], load_B_frag(wt, wstride, kc * 32, ntile), acc);
  return acc;
}

// Scatter a D-tile (C layout: lane l elem r -> M = r + (l<16?0:8), N = l%16)
// into the staging buffer as f16 for the next GEMM's A loads.
__device__ __forceinline__ void store_tile(_Float16* act, int ntile, v8f v) {
  const int l     = threadIdx.x & 31;
  const int col   = ntile * 16 + (l & 15);
  const int rbase = (l & 16) ? 8 : 0;
#pragma unroll
  for (int r = 0; r < 8; ++r)
    act[(rbase + r) * STR + col] = (_Float16)v[r];
}

// Layer with tanh activation stored to staging buffer.
// RANK1: add xr[row] * wlast[col] (the 129th GRU input row).
template <int KC, int NT, bool RANK1>
__device__ __forceinline__ void layer_tanh(const _Float16* src, _Float16* dst,
                                           const _Float16* wt, int wstride,
                                           const float* bias, const float* wlast,
                                           const float xr[8]) {
  const int c16 = threadIdx.x & 15;
  v16h af[KC];
  load_A_set<KC>(src, af);
  for (int j = 0; j < NT; ++j) {
    v8f acc = gemm_preA<KC>(af, wt, wstride, j);
    float bb = bias[j * 16 + c16];
    float wl = RANK1 ? wlast[j * 16 + c16] : 0.0f;
    v8f o;
#pragma unroll
    for (int r = 0; r < 8; ++r) {
      float v = acc[r] + bb;
      if (RANK1) v += xr[r] * wl;
      o[r] = fast_tanh(v);
    }
    store_tile(dst, j, o);
  }
}

// y_stage = y + c * k  ->  bufA cols 0..63 (A operand of ode_f)
__device__ __forceinline__ void write_stage(_Float16* bufA, const v8f y[4],
                                            const v8f k[4], float c) {
#pragma unroll
  for (int j = 0; j < 4; ++j) {
    v8f t;
#pragma unroll
    for (int r = 0; r < 8; ++r) t[r] = y[j][r] + c * k[j][r];
    store_tile(bufA, j, t);
  }
}

// k = W3 @ tanh(W2 @ tanh(W1 @ y + b1) + b2) + b3 for this wave's 16 rows.
__device__ __forceinline__ void ode_f(_Float16* bufA, _Float16* bufB,
                                      const _Float16* W1t, const _Float16* W2t,
                                      const _Float16* W3t, const float* b1,
                                      const float* b2, const float* b3,
                                      v8f k[4]) {
  const int c16 = threadIdx.x & 15;
  layer_tanh<2, 8, false>(bufA, bufB, W1t, 64, b1, nullptr, nullptr);   // K=64
  layer_tanh<4, 8, false>(bufB, bufA, W2t, 128, b2, nullptr, nullptr);  // K=128
  v16h af[4];
  load_A_set<4>(bufA, af);
#pragma unroll
  for (int j = 0; j < 4; ++j) {                 // k = h2@W3+b3, K=128, N=64
    v8f a = gemm_preA<4>(af, W3t, 128, j);
    float bb = b3[j * 16 + c16];
#pragma unroll
    for (int r = 0; r < 8; ++r) k[j][r] = a[r] + bb;
  }
}

__global__ __launch_bounds__(NTHREADS) void ode_rnn_wmma(
    const float* __restrict__ bin, const float* __restrict__ train_m,
    const float* __restrict__ W1, const float* __restrict__ b1g,
    const float* __restrict__ W2, const float* __restrict__ b2g,
    const float* __restrict__ W3, const float* __restrict__ b3g,
    const float* __restrict__ Wu1, const float* __restrict__ bu1g,
    const float* __restrict__ Wu2, const float* __restrict__ bu2g,
    const float* __restrict__ Wr1, const float* __restrict__ br1g,
    const float* __restrict__ Wr2, const float* __restrict__ br2g,
    const float* __restrict__ Wn1, const float* __restrict__ bn1g,
    const float* __restrict__ Wn2, const float* __restrict__ bn2g,
    float* __restrict__ out) {
  extern __shared__ char smem_raw[];
  _Float16* smh = (_Float16*)smem_raw;
  float*    smf = (float*)(smem_raw + (size_t)H_TOTAL * 2);

  _Float16* W1t  = smh + H_W1t;
  _Float16* W2t  = smh + H_W2t;
  _Float16* W3t  = smh + H_W3t;
  _Float16* Wu1t = smh + H_Wu1t;
  _Float16* Wu2t = smh + H_Wu2t;
  _Float16* Wr1t = smh + H_Wr1t;
  _Float16* Wr2t = smh + H_Wr2t;
  _Float16* Wn1t = smh + H_Wn1t;
  _Float16* Wn2t = smh + H_Wn2t;
  float *sb1 = smf + F_b1, *sb2 = smf + F_b2, *sb3 = smf + F_b3;
  float *sbu1 = smf + F_bu1, *sbu2 = smf + F_bu2;
  float *sbr1 = smf + F_br1, *sbr2 = smf + F_br2;
  float *sbn1 = smf + F_bn1, *sbn2 = smf + F_bn2;
  float *wu1l = smf + F_wu1l, *wr1l = smf + F_wr1l, *wn1l = smf + F_wn1l;

  const int tid = threadIdx.x;
  // One-time cooperative load: weights transposed to (N rows x K cols) f16.
  auto loadT = [&](_Float16* dst, const float* src, int K, int N) {
    for (int i = tid; i < K * N; i += NTHREADS) {
      int k = i / N, n = i - k * N;
      dst[n * K + k] = (_Float16)src[i];
    }
  };
  auto loadF = [&](float* dst, const float* src, int n) {
    for (int i = tid; i < n; i += NTHREADS) dst[i] = src[i];
  };
  loadT(W1t, W1, 64, 128);    loadT(W2t, W2, 128, 128);  loadT(W3t, W3, 128, 64);
  loadT(Wu1t, Wu1, 128, 128); loadT(Wu2t, Wu2, 128, 64);
  loadT(Wr1t, Wr1, 128, 128); loadT(Wr2t, Wr2, 128, 64);
  loadT(Wn1t, Wn1, 128, 128); loadT(Wn2t, Wn2, 128, 128);
  loadF(sb1, b1g, 128);  loadF(sb2, b2g, 128); loadF(sb3, b3g, 64);
  loadF(sbu1, bu1g, 128); loadF(sbu2, bu2g, 64);
  loadF(sbr1, br1g, 128); loadF(sbr2, br2g, 64);
  loadF(sbn1, bn1g, 128); loadF(sbn2, bn2g, 128);
  loadF(wu1l, Wu1 + 128 * 128, 128);   // 129th row (x weights), kept f32
  loadF(wr1l, Wr1 + 128 * 128, 128);
  loadF(wn1l, Wn1 + 128 * 128, 128);
  __syncthreads();

  const int wave  = tid >> 5;
  const int lane  = tid & 31;
  const int c16   = lane & 15;
  const int rbase = (lane & 16) ? 8 : 0;
  const int row0  = blockIdx.x * BTILE + wave * 16;

  _Float16* bufA = smh + H_STAGE + wave * (2 * 16 * STR);
  _Float16* bufB = bufA + 16 * STR;
  float* xw = smf + F_xls + wave * 16;
  float* mw = smf + F_mls + wave * 16;

  v8f y[4], s[4];
#pragma unroll
  for (int j = 0; j < 4; ++j)
#pragma unroll
    for (int r = 0; r < 8; ++r) { y[j][r] = 0.0f; s[j][r] = 0.0f; }

#pragma clang loop unroll(disable)
  for (int i = 0; i < T_; ++i) {
    const int tr = T_ - 1 - i;
    const float t1 = bin[tr * 2];
    const float t0 = (i == 0) ? THOR : bin[(tr + 1) * 2];

    if (lane < 16) {               // stage this wave's x / mask (intra-wave LDS)
      int grow = row0 + lane;
      xw[lane] = bin[(size_t)grow * (T_ * 2) + tr * 2 + 1];
      mw[lane] = train_m[(size_t)grow * T_ + tr];
    }
    float xr[8], mr[8];
#pragma unroll
    for (int r = 0; r < 8; ++r) { xr[r] = xw[rbase + r]; mr[r] = mw[rbase + r]; }

    const float hh = (t1 - t0) / (float)NSTEP;
#pragma clang loop unroll(disable)
    for (int st = 0; st < NSTEP; ++st) {      // ---- RK4 ----
      v8f k[4], ks[4];
      write_stage(bufA, y, y, 0.0f);
      ode_f(bufA, bufB, W1t, W2t, W3t, sb1, sb2, sb3, k);
#pragma unroll
      for (int j = 0; j < 4; ++j)
#pragma unroll
        for (int r = 0; r < 8; ++r) ks[j][r] = k[j][r];
      write_stage(bufA, y, k, 0.5f * hh);
      ode_f(bufA, bufB, W1t, W2t, W3t, sb1, sb2, sb3, k);
#pragma unroll
      for (int j = 0; j < 4; ++j)
#pragma unroll
        for (int r = 0; r < 8; ++r) ks[j][r] += 2.0f * k[j][r];
      write_stage(bufA, y, k, 0.5f * hh);
      ode_f(bufA, bufB, W1t, W2t, W3t, sb1, sb2, sb3, k);
#pragma unroll
      for (int j = 0; j < 4; ++j)
#pragma unroll
        for (int r = 0; r < 8; ++r) ks[j][r] += 2.0f * k[j][r];
      write_stage(bufA, y, k, hh);
      ode_f(bufA, bufB, W1t, W2t, W3t, sb1, sb2, sb3, k);
#pragma unroll
      for (int j = 0; j < 4; ++j)
#pragma unroll
        for (int r = 0; r < 8; ++r)
          y[j][r] += (hh * (1.0f / 6.0f)) * (ks[j][r] + k[j][r]);
    }

    // ---- GRU-Bayes update ----
#pragma unroll
    for (int j = 0; j < 4; ++j) { store_tile(bufA, j, y[j]); store_tile(bufA, j + 4, s[j]); }

    // update gate: u = sigmoid(tanh([y|s|x]@Wu1+bu1) @ Wu2 + bu2)
    layer_tanh<4, 8, true>(bufA, bufB, Wu1t, 128, sbu1, wu1l, xr);
    v8f u[4];
    {
      v16h af[4];
      load_A_set<4>(bufB, af);
#pragma unroll
      for (int j = 0; j < 4; ++j) {
        v8f a = gemm_preA<4>(af, Wu2t, 128, j);
        float bb = sbu2[j * 16 + c16];
#pragma unroll
        for (int r = 0; r < 8; ++r) u[j][r] = fast_sigmoid(a[r] + bb);
      }
    }
    // reset gate
    layer_tanh<4, 8, true>(bufA, bufB, Wr1t, 128, sbr1, wr1l, xr);
    v8f rg[4];
    {
      v16h af[4];
      load_A_set<4>(bufB, af);
#pragma unroll
      for (int j = 0; j < 4; ++j) {
        v8f a = gemm_preA<4>(af, Wr2t, 128, j);
        float bb = sbr2[j * 16 + c16];
#pragma unroll
        for (int r = 0; r < 8; ++r) rg[j][r] = fast_sigmoid(a[r] + bb);
      }
    }
    // yc = [mean*r | std*r] (+x rank-1)
#pragma unroll
    for (int j = 0; j < 4; ++j) {
      v8f tm, ts;
#pragma unroll
      for (int r = 0; r < 8; ++r) { tm[r] = y[j][r] * rg[j][r]; ts[r] = s[j][r] * rg[j][r]; }
      store_tile(bufA, j, tm);
      store_tile(bufA, j + 4, ts);
    }
    // candidate: ns = tanh(yc@Wn1+bn1) @ Wn2 + bn2 ; blend + mask
    layer_tanh<4, 8, true>(bufA, bufB, Wn1t, 128, sbn1, wn1l, xr);
    {
      v16h af[4];
      load_A_set<4>(bufB, af);
#pragma unroll
      for (int j = 0; j < 8; ++j) {
        v8f a = gemm_preA<4>(af, Wn2t, 128, j);
        float bb = sbn2[j * 16 + c16];
        if (j < 4) {
#pragma unroll
          for (int r = 0; r < 8; ++r) {
            float ns = a[r] + bb;
            float nm = (1.0f - u[j][r]) * ns + u[j][r] * y[j][r];
            y[j][r] = mr[r] * nm + (1.0f - mr[r]) * y[j][r];
          }
        } else {
          const int q = j - 4;
#pragma unroll
          for (int r = 0; r < 8; ++r) {
            float ns = a[r] + bb;
            float nv = (1.0f - u[q][r]) * ns + u[q][r] * s[q][r];
            float fv = mr[r] * nv + (1.0f - mr[r]) * s[q][r];
            s[q][r] = fabsf(fv);
          }
        }
      }
    }
  }

  // ---- write (mean, std) concatenated flat ----
  float* out_mean = out;
  float* out_std  = out + (size_t)B_ * LO_;
#pragma unroll
  for (int j = 0; j < 4; ++j)
#pragma unroll
    for (int r = 0; r < 8; ++r) {
      const size_t row = (size_t)(row0 + rbase + r);
      out_mean[row * LO_ + j * 16 + c16] = y[j][r];
      out_std [row * LO_ + j * 16 + c16] = s[j][r];
    }
}

extern "C" void kernel_launch(void* const* d_in, const int* in_sizes, int n_in,
                              void* d_out, int out_size, void* d_ws, size_t ws_size,
                              hipStream_t stream) {
  const float* bin     = (const float*)d_in[0];
  const float* train_m = (const float*)d_in[1];
  const float* W1  = (const float*)d_in[2];
  const float* b1  = (const float*)d_in[3];
  const float* W2  = (const float*)d_in[4];
  const float* b2  = (const float*)d_in[5];
  const float* W3  = (const float*)d_in[6];
  const float* b3  = (const float*)d_in[7];
  const float* Wu1 = (const float*)d_in[8];
  const float* bu1 = (const float*)d_in[9];
  const float* Wu2 = (const float*)d_in[10];
  const float* bu2 = (const float*)d_in[11];
  const float* Wr1 = (const float*)d_in[12];
  const float* br1 = (const float*)d_in[13];
  const float* Wr2 = (const float*)d_in[14];
  const float* br2 = (const float*)d_in[15];
  const float* Wn1 = (const float*)d_in[16];
  const float* bn1 = (const float*)d_in[17];
  const float* Wn2 = (const float*)d_in[18];
  const float* bn2 = (const float*)d_in[19];
  float* out = (float*)d_out;

  dim3 grid(B_ / BTILE), block(NTHREADS);
  ode_rnn_wmma<<<grid, block, SMEM_BYTES, stream>>>(
      bin, train_m, W1, b1, W2, b2, W3, b3, Wu1, bu1, Wu2, bu2,
      Wr1, br1, Wr2, br2, Wn1, bn1, Wn2, bn2, out);
}
